// FeedForward_69544110457391
// MI455X (gfx1250) — 8-bit matrix operand order repaired and checked by reading, compiled, not run on hardware
//
#include <hip/hip_runtime.h>
#include <stdint.h>
#include <stddef.h>

typedef int v8i __attribute__((ext_vector_type(8)));
typedef signed char i8;

#define QEPS 1e-5f

static constexpr int R_TOK = 2 * 4096;
static constexpr int DIM_D = 2048;
static constexpr int DIM_H = 8192;

static constexpr size_t OFF_W1Q  = 0;
static constexpr size_t OFF_W2Q  = OFF_W1Q + (size_t)DIM_H * DIM_D;
static constexpr size_t OFF_XQ   = OFF_W2Q + (size_t)DIM_D * DIM_H;
static constexpr size_t OFF_HQ   = OFF_XQ  + (size_t)R_TOK * DIM_D;
static constexpr size_t OFF_HBUF = OFF_HQ  + (size_t)R_TOK * DIM_H;
static constexpr size_t OFF_DQX  = OFF_HBUF + (size_t)R_TOK * DIM_H * 2;
static constexpr size_t OFF_DQH  = OFF_DQX + (size_t)R_TOK * 4;
static constexpr size_t OFF_WSUM = OFF_DQH + (size_t)R_TOK * 4;

__global__ __launch_bounds__(256) void bitnet_abssum(const float* __restrict__ w,
                                                     size_t n, float* __restrict__ slot) {
  __shared__ float red[256];
  float s = 0.f;
  size_t stride = (size_t)gridDim.x * blockDim.x;
  for (size_t i = (size_t)blockIdx.x * blockDim.x + threadIdx.x; i < n; i += stride)
    s += fabsf(w[i]);
  red[threadIdx.x] = s;
  __syncthreads();
  for (int off = 128; off > 0; off >>= 1) {
    if ((int)threadIdx.x < off) red[threadIdx.x] += red[threadIdx.x + off];
    __syncthreads();
  }
  if (threadIdx.x == 0) atomicAdd(slot, red[0]);
}

__global__ __launch_bounds__(256) void bitnet_quantw(const float* __restrict__ w,
                                                     i8* __restrict__ wq, size_t n4,
                                                     const float* __restrict__ slot,
                                                     float inv_n) {
  const float mean  = fmaxf(slot[0] * inv_n, QEPS);
  const float scale = 1.f / mean;
  const float4* w4 = (const float4*)w;
  int* out = (int*)wq;
  size_t stride = (size_t)gridDim.x * blockDim.x;
  for (size_t i = (size_t)blockIdx.x * blockDim.x + threadIdx.x; i < n4; i += stride) {
    float4 v = w4[i];
    int q0 = (int)fminf(fmaxf(rintf(v.x * scale), -1.f), 1.f);
    int q1 = (int)fminf(fmaxf(rintf(v.y * scale), -1.f), 1.f);
    int q2 = (int)fminf(fmaxf(rintf(v.z * scale), -1.f), 1.f);
    int q3 = (int)fminf(fmaxf(rintf(v.w * scale), -1.f), 1.f);
    out[i] = (q0 & 0xff) | ((q1 & 0xff) << 8) | ((q2 & 0xff) << 16) | ((q3 & 0xff) << 24);
  }
}

__global__ __launch_bounds__(256) void bitnet_quantx(const float* __restrict__ x,
                                                     i8* __restrict__ xq,
                                                     float* __restrict__ dqx) {
  __shared__ float red[256];
  const int row = blockIdx.x;
  const int t = threadIdx.x;
  const float4* x4 = (const float4*)(x + (size_t)row * DIM_D);
  float4 a = x4[t * 2 + 0];
  float4 b = x4[t * 2 + 1];
  float v[8] = {a.x, a.y, a.z, a.w, b.x, b.y, b.z, b.w};
  float m = 0.f;
#pragma unroll
  for (int i = 0; i < 8; ++i) m = fmaxf(m, fabsf(v[i]));
  red[t] = m;
  __syncthreads();
  for (int off = 128; off > 0; off >>= 1) {
    if (t < off) red[t] = fmaxf(red[t], red[t + off]);
    __syncthreads();
  }
  const float amax  = fmaxf(red[0], QEPS);
  const float scale = 127.f / amax;
  if (t == 0) dqx[row] = amax / 127.f;
  unsigned int lo = 0, hi = 0;
#pragma unroll
  for (int i = 0; i < 4; ++i) {
    int q = (int)fminf(fmaxf(rintf(v[i] * scale), -128.f), 127.f);
    lo |= (unsigned)(q & 0xff) << (8 * i);
  }
#pragma unroll
  for (int i = 0; i < 4; ++i) {
    int q = (int)fminf(fmaxf(rintf(v[4 + i] * scale), -128.f), 127.f);
    hi |= (unsigned)(q & 0xff) << (8 * i);
  }
  uint2 pk; pk.x = lo; pk.y = hi;
  *(uint2*)(xq + (size_t)row * DIM_D + t * 8) = pk;
}

__global__ __launch_bounds__(256) void bitnet_quanth(const _Float16* __restrict__ h,
                                                     i8* __restrict__ hq,
                                                     float* __restrict__ dqh) {
  __shared__ float red[256];
  const int row = blockIdx.x;
  const int t = threadIdx.x;
  const _Float16* hr = h + (size_t)row * DIM_H + t * 32;
  float v[32];
#pragma unroll
  for (int i = 0; i < 32; ++i) v[i] = (float)hr[i];
  float m = 0.f;
#pragma unroll
  for (int i = 0; i < 32; ++i) m = fmaxf(m, fabsf(v[i]));
  red[t] = m;
  __syncthreads();
  for (int off = 128; off > 0; off >>= 1) {
    if (t < off) red[t] = fmaxf(red[t], red[t + off]);
    __syncthreads();
  }
  const float amax  = fmaxf(red[0], QEPS);
  const float scale = 127.f / amax;
  if (t == 0) dqh[row] = amax / 127.f;
  int q[8];
#pragma unroll
  for (int j = 0; j < 8; ++j) {
    int p = 0;
#pragma unroll
    for (int i = 0; i < 4; ++i) {
      int qq = (int)fminf(fmaxf(rintf(v[j * 4 + i] * scale), -128.f), 127.f);
      p |= (qq & 0xff) << (8 * i);
    }
    q[j] = p;
  }
  int4* dst = (int4*)(hq + (size_t)row * DIM_H + t * 32);
  dst[0] = make_int4(q[0], q[1], q[2], q[3]);
  dst[1] = make_int4(q[4], q[5], q[6], q[7]);
}

template <bool RELU_F16_OUT>
__global__ __launch_bounds__(256) void bitnet_gemm(const i8* __restrict__ A,
                                                   const i8* __restrict__ B,
                                                   const float* __restrict__ dq_row,
                                                   const float* __restrict__ wsum,
                                                   float winv_n,
                                                   const float* __restrict__ bias,
                                                   void* __restrict__ out,
                                                   int K, int Nout) {
  constexpr int TILE_BYTES = 128 * 64;
  __shared__ __align__(16) i8 As[2][TILE_BYTES];
  __shared__ __align__(16) i8 Bs[2][TILE_BYTES];

  const int t    = threadIdx.x;
  const int lane = t & 31;
  const int wave = t >> 5;
  const int wm   = wave >> 1;
  const int wn   = wave & 1;
  const int hi   = lane >> 4;
  const int ln   = lane & 15;

  const int rowBase = blockIdx.y * 128;
  const int colBase = blockIdx.x * 128;

  const float dq_w = fmaxf(wsum[0] * winv_n, QEPS);

  const uint32_t ldsA = (uint32_t)(uintptr_t)(void*)&As[0][0];
  const uint32_t ldsB = (uint32_t)(uintptr_t)(void*)&Bs[0][0];

  auto stage = [&](int buf, int k0) {
#pragma unroll
    for (int s = 0; s < 2; ++s) {
      const int c  = t + s * 256;
      const int r  = c >> 2;
      const int ko = (c & 3) * 16;
      const uint64_t gA = (uint64_t)(uintptr_t)(A + (size_t)(rowBase + r) * K + k0 + ko);
      const uint64_t gB = (uint64_t)(uintptr_t)(B + (size_t)(colBase + r) * K + k0 + ko);
      const uint32_t lA = ldsA + (uint32_t)(buf * TILE_BYTES + c * 16);
      const uint32_t lB = ldsB + (uint32_t)(buf * TILE_BYTES + c * 16);
      asm volatile("global_load_async_to_lds_b128 %0, %1, off"
                   :: "v"(lA), "v"(gA) : "memory");
      asm volatile("global_load_async_to_lds_b128 %0, %1, off"
                   :: "v"(lB), "v"(gB) : "memory");
    }
  };

  v8i acc[2][4];
#pragma unroll
  for (int mt = 0; mt < 2; ++mt)
#pragma unroll
    for (int nt = 0; nt < 4; ++nt) acc[mt][nt] = (v8i){};

  stage(0, 0);
  asm volatile("s_wait_asynccnt 0x0" ::: "memory");
  __syncthreads();

  int cur = 0;
  for (int k0 = 0; k0 < K; k0 += 64) {
    const int nxt = cur ^ 1;
    if (k0 + 64 < K) stage(nxt, k0 + 64);

    const i8* __restrict__ Atile = &As[cur][0];
    const i8* __restrict__ Btile = &Bs[cur][0];

    v8i af[2];
#pragma unroll
    for (int mt = 0; mt < 2; ++mt) {
      const int rloc = wm * 32 + mt * 16 + ln;
#pragma unroll
      for (int v = 0; v < 8; ++v) {
        const int k = ((v >> 1) << 4) + (hi << 3) + ((v & 1) << 2);
        af[mt][v] = *(const int*)&Atile[rloc * 64 + k];
      }
    }
#pragma unroll
    for (int nt = 0; nt < 4; ++nt) {
      const int cloc = wn * 64 + nt * 16 + ln;
      v8i bf;
#pragma unroll
      for (int v = 0; v < 8; ++v) {
        const int k = ((v >> 1) << 4) + (hi << 3) + ((v & 1) << 2);
        bf[v] = *(const int*)&Btile[cloc * 64 + k];
      }
#pragma unroll
      for (int mt = 0; mt < 2; ++mt) {
        acc[mt][nt] = __builtin_amdgcn_wmma_i32_16x16x64_iu8(
            true, af[mt], true, bf, acc[mt][nt], false, false);
      }
    }

    asm volatile("s_wait_asynccnt 0x0" ::: "memory");
    __syncthreads();
    cur = nxt;
  }

#pragma unroll
  for (int mt = 0; mt < 2; ++mt) {
#pragma unroll
    for (int nt = 0; nt < 4; ++nt) {
      const int col = colBase + wn * 64 + nt * 16 + ln;
      const float bv = bias[col];
#pragma unroll
      for (int r = 0; r < 8; ++r) {
        const int row = rowBase + wm * 32 + mt * 16 + r + hi * 8;
        float val = (float)acc[mt][nt][r] * dq_row[row] * dq_w + bv;
        if (RELU_F16_OUT) {
          val = fmaxf(val, 0.f);
          ((_Float16*)out)[(size_t)row * Nout + col] = (_Float16)val;
        } else {
          ((float*)out)[(size_t)row * Nout + col] = val;
        }
      }
    }
  }
}

extern "C" void kernel_launch(void* const* d_in, const int* in_sizes, int n_in,
                              void* d_out, int out_size, void* d_ws, size_t ws_size,
                              hipStream_t stream) {
  (void)in_sizes; (void)n_in; (void)out_size; (void)ws_size;
  const float* x  = (const float*)d_in[0];
  const float* w1 = (const float*)d_in[1];
  const float* b1 = (const float*)d_in[2];
  const float* w2 = (const float*)d_in[3];
  const float* b2 = (const float*)d_in[4];
  float* out = (float*)d_out;

  char* ws = (char*)d_ws;
  i8*       w1q  = (i8*)(ws + OFF_W1Q);
  i8*       w2q  = (i8*)(ws + OFF_W2Q);
  i8*       xq   = (i8*)(ws + OFF_XQ);
  i8*       hq   = (i8*)(ws + OFF_HQ);
  _Float16* hbuf = (_Float16*)(ws + OFF_HBUF);
  float*    dqx  = (float*)(ws + OFF_DQX);
  float*    dqh  = (float*)(ws + OFF_DQH);
  float*    wsum = (float*)(ws + OFF_WSUM);

  const size_t nW = (size_t)DIM_H * DIM_D;
  const float  winv = 1.f / (float)nW;

  hipMemsetAsync(wsum, 0, 2 * sizeof(float), stream);

  bitnet_abssum<<<1024, 256, 0, stream>>>(w1, nW, wsum + 0);
  bitnet_abssum<<<1024, 256, 0, stream>>>(w2, nW, wsum + 1);
  bitnet_quantw<<<2048, 256, 0, stream>>>(w1, w1q, nW / 4, wsum + 0, winv);
  bitnet_quantw<<<2048, 256, 0, stream>>>(w2, w2q, nW / 4, wsum + 1, winv);
  bitnet_quantx<<<R_TOK, 256, 0, stream>>>(x, xq, dqx);

  dim3 g1(DIM_H / 128, R_TOK / 128);
  bitnet_gemm<true><<<g1, 256, 0, stream>>>(xq, w1q, dqx, wsum + 0, winv, b1,
                                            (void*)hbuf, DIM_D, DIM_H);

  bitnet_quanth<<<R_TOK, 256, 0, stream>>>(hbuf, hq, dqh);

  dim3 g2(DIM_D / 128, R_TOK / 128);
  bitnet_gemm<false><<<g2, 256, 0, stream>>>(hq, w2q, dqh, wsum + 1, winv, b2,
                                             (void*)out, DIM_H, DIM_D);
}
